// DynamicGraphCNN_60619168416175
// MI455X (gfx1250) — compile-verified, matrix operand order checked by reading, not run on hardware
//
#include <hip/hip_runtime.h>

// ---------------------------------------------------------------------------
// DGCNN (3x EdgeConv + global max + linear head) for MI455X (gfx1250).
// - bf16 WMMA (v_wmma_f32_16x16x32_bf16) for pairwise-Gram (kNN) and edge-MLP.
// - kNN column tiles staged into LDS with the Tensor Data Mover
//   (tensor_load_to_lds + s_wait_tensorcnt), double-buffered.
// - Distance matrix never materialized in HBM; BN+ReLU+max-over-k fused via
//   the monotone-affine trick (2-pass BN stats).
// B=8, N=2048, k=20 hardcoded per the reference setup.
// ---------------------------------------------------------------------------

typedef __attribute__((ext_vector_type(16))) __bf16 v16bf;
typedef __attribute__((ext_vector_type(8)))  __bf16 v8bf;
typedef __attribute__((ext_vector_type(8)))  float  v8f;
typedef __attribute__((ext_vector_type(4)))  unsigned uint32x4;
typedef __attribute__((ext_vector_type(8)))  int      int32x8;
typedef __attribute__((ext_vector_type(4)))  int      int32x4;

#define BATCH 8
#define NPTS  2048
#define KNN   20
#define KPAD  32

// ---- order-preserving float <-> u32 map (for atomic max on floats) --------
__device__ __forceinline__ unsigned fmap(float f) {
  int b = __float_as_int(f);
  return (unsigned)(b ^ ((b >> 31) | 0x80000000));
}
__device__ __forceinline__ float funmap(unsigned u) {
  int b = (int)(u ^ (((int)u < 0) ? 0x80000000u : 0xffffffffu));
  return __int_as_float(b);
}

// ---- LDS byte offset of a __shared__ object (low 32 bits of flat addr) ----
__device__ __forceinline__ unsigned ldsoff(const void* p) {
  return (unsigned)(unsigned long long)(uintptr_t)p;
}

// ---- load one lane's A/B fragment (16-bit 16x32 layout) -------------------
// lane m holds row M; K chunks {kb..kb+7} -> VGPR0..3, {kb+16..kb+23} -> 4..7
__device__ __forceinline__ v16bf frag_ab(const __bf16* __restrict__ row, int kb) {
  v16bf f;
  v8bf lo = *reinterpret_cast<const v8bf*>(row + kb);
  v8bf hi = *reinterpret_cast<const v8bf*>(row + kb + 16);
#pragma unroll
  for (int i = 0; i < 8; ++i) { f[i] = lo[i]; f[8 + i] = hi[i]; }
  return f;
}

// ---------------------------------------------------------------------------
// TDM: 2D tile load (rows x row_elems bf16, row stride = stride_elems) from
// global memory into LDS.  D# packed per CDNA5 ISA 8.3/8.4:
//   group0: count=1 | lds_addr | global_addr(57b) | type=2
//   group1: data_size=1 (2B) | tensor_dim0/1 | tile_dim0/1 | dim0 stride
// Issued once per wave (EXEC-independent); tracked by TENSORcnt.
// ---------------------------------------------------------------------------
__device__ __forceinline__ void tdm_load_2d(unsigned lds_addr, const void* gptr,
                                            unsigned rows, unsigned row_elems,
                                            unsigned stride_elems,
                                            unsigned tensor_rows) {
  unsigned long long ga = (unsigned long long)(uintptr_t)gptr;
  uint32x4 g0;
  g0[0] = 1u;                                             // count=1 (valid)
  g0[1] = lds_addr;                                       // lds_addr
  g0[2] = (unsigned)(ga & 0xffffffffu);                   // global_addr[31:0]
  g0[3] = (unsigned)((ga >> 32) & 0x01ffffffu) | (2u << 30); // [56:32] | type=2
  int32x8 g1;
  g1[0] = (int)(1u << 16);                                // data_size=1 (2B)
  g1[1] = (int)((row_elems & 0xffffu) << 16);             // tensor_dim0 lo16
  g1[2] = (int)((row_elems >> 16) | ((tensor_rows & 0xffffu) << 16));
  g1[3] = (int)((tensor_rows >> 16) | ((row_elems & 0xffffu) << 16)); // tile_dim0
  g1[4] = (int)(rows & 0xffffu);                          // tile_dim1 (tile_dim2=0)
  g1[5] = (int)stride_elems;                              // tensor_dim0_stride lo32
  g1[6] = 0;                                              // stride hi16 / dim1 stride
  g1[7] = 0;
  int32x4 z4 = {0, 0, 0, 0};
  int32x8 z8 = {0, 0, 0, 0, 0, 0, 0, 0};
  __builtin_amdgcn_tensor_load_to_lds(g0, g1, z4, z4, z8, 0);
}

// ===========================================================================
// prep0: x (B,N,3) fp32 -> xb0 (B,N,32) bf16 zero-padded + row norms (bf16-
// rounded for consistency with the WMMA Gram products).
// ===========================================================================
__global__ void prep0_kernel(const float* __restrict__ x, __bf16* __restrict__ xb,
                             float* __restrict__ xx) {
  int i = blockIdx.x * blockDim.x + threadIdx.x;
  if (i >= BATCH * NPTS) return;
  __bf16 a = (__bf16)x[i * 3 + 0];
  __bf16 b = (__bf16)x[i * 3 + 1];
  __bf16 c = (__bf16)x[i * 3 + 2];
  __bf16* p = xb + (size_t)i * 32;
  p[0] = a; p[1] = b; p[2] = c;
#pragma unroll
  for (int j = 3; j < 32; ++j) p[j] = (__bf16)0.0f;
  float fa = (float)a, fb = (float)b, fc = (float)c;
  xx[i] = fa * fa + fb * fb + fc * fc;
}

// ===========================================================================
// wprep: W (O, 2*CIN) fp32 -> bf16 (O, KP) with each half zero-padded to KP/2
// ===========================================================================
template <int O, int CIN, int KP>
__global__ void wprep_kernel(const float* __restrict__ w, __bf16* __restrict__ wb) {
  int i = blockIdx.x * blockDim.x + threadIdx.x;
  if (i >= O * KP) return;
  int o = i / KP, k = i % KP;
  float v = 0.0f;
  constexpr int H = KP / 2;
  if (k < H) { if (k < CIN) v = w[o * 2 * CIN + k]; }
  else       { int c = k - H; if (c < CIN) v = w[o * 2 * CIN + CIN + c]; }
  wb[i] = (__bf16)v;
}

// ===========================================================================
// zero: clear fp32 / u32 buffers (graph-capture-safe; no hipMemset)
// ===========================================================================
__global__ void zero_kernel(unsigned* __restrict__ p, int n) {
  int i = blockIdx.x * blockDim.x + threadIdx.x;
  if (i < n) p[i] = 0u;
}

// ===========================================================================
// knn: fused WMMA Gram + top-k.  Block = 64 threads (2 waves) owns 64 rows.
// Columns streamed in chunks of 64, TDM double-buffered into LDS; the 4x4
// grid of 16x16 WMMA tiles per chunk is split across the 2 waves, neg_dist
// goes to LDS, then each thread keeps a register-resident sorted top-20.
// ===========================================================================
template <int CP>
__global__ __launch_bounds__(64) void knn_kernel(const __bf16* __restrict__ xb,
                                                 const float* __restrict__ xx,
                                                 int* __restrict__ idx_out) {
  constexpr int ROWS = 64, CCH = 64;
  __shared__ float  dist[ROWS * CCH];        // 16 KB
  __shared__ __bf16 colbuf[2][CCH * CP];     // <= 32 KB (CP=128)
  __shared__ float  xxr[ROWS];
  __shared__ float  xxc[CCH];

  const int b  = blockIdx.y;
  const int r0 = blockIdx.x * ROWS;
  const __bf16* xbase = xb + (size_t)b * NPTS * CP;
  const float*  xxb   = xx + (size_t)b * NPTS;

  const int t = threadIdx.x;
  const int wave = t >> 5, lane = t & 31;
  const int m = lane & 15, half = lane >> 4;

  float val[KNN];
  int   ind[KNN];
#pragma unroll
  for (int i = 0; i < KNN; ++i) { val[i] = -3.4e38f; ind[i] = 0; }

  xxr[t] = xxb[r0 + t];

  // prime the pipeline: TDM chunk 0 -> buffer 0 (wave 0 only)
  if (wave == 0) tdm_load_2d(ldsoff(&colbuf[0][0]), xbase, CCH, CP, CP, NPTS);

  int buf = 0;
  for (int c0 = 0; c0 < NPTS; c0 += CCH) {
    if (wave == 0) {
      if (c0 + CCH < NPTS) {
        tdm_load_2d(ldsoff(&colbuf[buf ^ 1][0]),
                    xbase + (size_t)(c0 + CCH) * CP, CCH, CP, CP, NPTS - (c0 + CCH));
        __builtin_amdgcn_s_wait_tensorcnt(1);  // chunk c done, c+1 in flight
      } else {
        __builtin_amdgcn_s_wait_tensorcnt(0);
      }
    }
    for (int j = t; j < CCH; j += 64) xxc[j] = xxb[c0 + j];
    __syncthreads();

    // ---- WMMA fill of the 64x64 neg_dist tile ----
    for (int tile = wave; tile < 16; tile += 2) {
      const int rt = tile & 3, ct = tile >> 2;
      const __bf16* arow = xbase + (size_t)(r0 + rt * 16 + m) * CP;
      const __bf16* brow = &colbuf[buf][(ct * 16 + m) * CP];
      v8f acc = {};
#pragma unroll
      for (int k0 = 0; k0 < CP; k0 += 32) {
        v16bf a  = frag_ab(arow, k0 + half * 8);
        v16bf bb = frag_ab(brow, k0 + half * 8);
        acc = __builtin_amdgcn_wmma_f32_16x16x32_bf16(false, a, false, bb,
                                                      (short)0, acc, false, false);
      }
      const float xc = xxc[ct * 16 + m];
#pragma unroll
      for (int v = 0; v < 8; ++v) {
        const int rr = rt * 16 + v + 8 * half;           // C/D row layout
        dist[rr * CCH + ct * 16 + m] = 2.0f * acc[v] - xxr[rr] - xc;
      }
    }
    __syncthreads();

    // ---- per-row top-20 maintenance (registers, unrolled swap cascade) ----
    {
      const float* drow = &dist[t * CCH];
      for (int j = 0; j < CCH; ++j) {
        const float v = drow[j];
        if (v > val[KNN - 1]) {
          float cv = v; int ci = c0 + j;
#pragma unroll
          for (int s = 0; s < KNN; ++s) {
            if (cv > val[s]) {
              float tv = val[s]; val[s] = cv; cv = tv;
              int   ti = ind[s]; ind[s] = ci; ci = ti;
            }
          }
        }
      }
    }
    __syncthreads();   // dist + colbuf[buf] free for reuse
    buf ^= 1;
  }

  int* op = idx_out + ((size_t)b * NPTS + r0 + t) * KPAD;
#pragma unroll
  for (int i = 0; i < KNN; ++i) op[i] = ind[i];
#pragma unroll
  for (int i = KNN; i < KPAD; ++i) op[i] = ind[0];   // pad: duplicate (max-safe)
}

// ===========================================================================
// mlp: one block (8 waves) per point. Builds edge features (32 x KP) bf16 in
// LDS, runs 2 x (O/16) WMMA 16x16 tile jobs across the waves, fuses bias add,
// per-channel max over k (pre-BN, monotone-affine trick) and BN-stat
// accumulation (rows kk<20 only).
// ===========================================================================
template <int CIN, int KP, int O>
__global__ __launch_bounds__(256) void mlp_kernel(const float* __restrict__ x,
                                                  const int* __restrict__ idx,
                                                  const __bf16* __restrict__ wb,
                                                  const float* __restrict__ bias,
                                                  float* __restrict__ hmax,
                                                  float* __restrict__ gstats) {
  __shared__ __bf16   ef[KPAD * KP];
  __shared__ float    ssum[O];
  __shared__ float    ssq[O];
  __shared__ unsigned smax[O];

  const int t = threadIdx.x, wave = t >> 5, lane = t & 31;
  const int bn = blockIdx.x;
  const int b = bn / NPTS, n = bn % NPTS;
  const float* xb_ = x + (size_t)b * NPTS * CIN;
  const float* ctr = xb_ + (size_t)n * CIN;
  const int*   ip  = idx + (size_t)bn * KPAD;

  // warm B-matrix into cache (global_prefetch_b8, one line per 128B)
  for (int p = t * 64; p < O * KP; p += 256 * 64) __builtin_prefetch(wb + p, 0, 1);

  for (int j = t; j < O; j += 256) { ssum[j] = 0.f; ssq[j] = 0.f; smax[j] = 0u; }

  // ---- build edge features: [ctr | nbr - ctr], each half padded to KP/2 ----
  for (int e = t; e < KPAD * CIN; e += 256) {
    const int kk = e / CIN, c = e % CIN;
    const int nb = ip[kk];
    const float cv = ctr[c];
    const float nv = xb_[(size_t)nb * CIN + c];
    ef[kk * KP + c]            = (__bf16)cv;
    ef[kk * KP + (KP / 2) + c] = (__bf16)(nv - cv);
  }
  if constexpr (2 * CIN < KP) {
    constexpr int PADC = KP / 2 - CIN;
    for (int e = t; e < KPAD * PADC; e += 256) {
      const int kk = e / PADC, c = CIN + e % PADC;
      ef[kk * KP + c]            = (__bf16)0.0f;
      ef[kk * KP + (KP / 2) + c] = (__bf16)0.0f;
    }
  }
  __syncthreads();

  // ---- GEMM: (32 x KP) x (KP x O), 16x16 WMMA tiles ----
  constexpr int JOBS = 2 * (O / 16);
  const int m = lane & 15, half = lane >> 4;
  for (int job = wave; job < JOBS; job += 8) {
    const int mt = job & 1, ct = job >> 1;
    const __bf16* arow = &ef[(mt * 16 + m) * KP];
    const __bf16* brow = wb + (size_t)(ct * 16 + m) * KP;
    v8f acc = {};
#pragma unroll
    for (int k0 = 0; k0 < KP; k0 += 32) {
      v16bf a  = frag_ab(arow, k0 + half * 8);
      v16bf bb = frag_ab(brow, k0 + half * 8);
      acc = __builtin_amdgcn_wmma_f32_16x16x32_bf16(false, a, false, bb,
                                                    (short)0, acc, false, false);
    }
    const int oc = ct * 16 + m;
    const float bv = bias[oc];
    float s = 0.f, s2 = 0.f;
    unsigned mx = 0u;
#pragma unroll
    for (int v = 0; v < 8; ++v) {
      const int kk = mt * 16 + v + 8 * half;
      const float h = acc[v] + bv;
      if (kk < KNN) { s += h; s2 += h * h; }
      const unsigned u = fmap(h);
      mx = mx > u ? mx : u;
    }
    atomicAdd(&ssum[oc], s);
    atomicAdd(&ssq[oc], s2);
    atomicMax(&smax[oc], mx);
  }
  __syncthreads();

  for (int j = t; j < O; j += 256) {
    hmax[(size_t)bn * O + j] = funmap(smax[j]);
    atomicAdd(&gstats[j],     ssum[j]);
    atomicAdd(&gstats[O + j], ssq[j]);
  }
}

// ===========================================================================
// bnfin: finalize BN (scale>0 => max commutes) + ReLU. Non-last layers emit
// next-layer fp32 / bf16 activations + row norms; last layer atomic-maxes the
// global feature over N.
// ===========================================================================
template <int O, bool LAST>
__global__ void bnfin_kernel(const float* __restrict__ hmax,
                             const float* __restrict__ gstats,
                             const float* __restrict__ gamma,
                             const float* __restrict__ beta,
                             float* __restrict__ xnext,
                             __bf16* __restrict__ xbnext,
                             float* __restrict__ xxnext,
                             unsigned* __restrict__ gfeat, float invcnt) {
  __shared__ float red[O];
  const int bn = blockIdx.x, t = threadIdx.x;   // blockDim == O
  const float mean = gstats[t] * invcnt;
  const float var  = gstats[O + t] * invcnt - mean * mean;
  const float sc = gamma[t] * rsqrtf(var + 1e-5f);
  const float sh = beta[t] - mean * sc;
  const float y = fmaxf(sc * hmax[(size_t)bn * O + t] + sh, 0.0f);
  if constexpr (LAST) {
    atomicMax(&gfeat[(bn / NPTS) * O + t], fmap(y));
  } else {
    const __bf16 yb = (__bf16)y;
    xnext[(size_t)bn * O + t]  = y;
    xbnext[(size_t)bn * O + t] = yb;
    const float yf = (float)yb;
    red[t] = yf * yf;
    __syncthreads();
    for (int s = O / 2; s > 0; s >>= 1) {
      if (t < s) red[t] += red[t + s];
      __syncthreads();
    }
    if (t == 0) xxnext[bn] = red[0];
  }
}

// ===========================================================================
// head: out[b,o] = sum_c gfeat[b,c]*wo[o,c] + bo[o]   (8x256, tiny)
// ===========================================================================
__global__ void head_kernel(const unsigned* __restrict__ gfeat,
                            const float* __restrict__ wo,
                            const float* __restrict__ bo,
                            float* __restrict__ out) {
  const int i = blockIdx.x * blockDim.x + threadIdx.x;
  if (i >= BATCH * 256) return;
  const int b = i / 256, o = i % 256;
  const unsigned* g = gfeat + b * 256;
  float acc = bo[o];
  for (int c = 0; c < 256; ++c) acc += funmap(g[c]) * wo[o * 256 + c];
  out[i] = acc;
}

// ===========================================================================
extern "C" void kernel_launch(void* const* d_in, const int* in_sizes, int n_in,
                              void* d_out, int out_size, void* d_ws, size_t ws_size,
                              hipStream_t stream) {
  const float* x   = (const float*)d_in[0];
  const float* w1  = (const float*)d_in[1];
  const float* b1  = (const float*)d_in[2];
  const float* g1  = (const float*)d_in[3];
  const float* be1 = (const float*)d_in[4];
  const float* w2  = (const float*)d_in[5];
  const float* b2  = (const float*)d_in[6];
  const float* g2  = (const float*)d_in[7];
  const float* be2 = (const float*)d_in[8];
  const float* w3  = (const float*)d_in[9];
  const float* b3  = (const float*)d_in[10];
  const float* g3  = (const float*)d_in[11];
  const float* be3 = (const float*)d_in[12];
  const float* wo  = (const float*)d_in[13];
  const float* bo  = (const float*)d_in[14];
  float* out = (float*)d_out;

  const size_t BN = (size_t)BATCH * NPTS;
  char* ws = (char*)d_ws;
  size_t off = 0;
  auto alloc = [&](size_t bytes) { void* p = ws + off; off = (off + bytes + 255) & ~(size_t)255; return p; };

  __bf16* xb0   = (__bf16*)alloc(BN * 32 * 2);
  float*  xx0   = (float*) alloc(BN * 4);
  int*    idxb  = (int*)   alloc(BN * KPAD * 4);
  float*  hmax  = (float*) alloc(BN * 256 * 4);
  float*  stats = (float*) alloc(512 * 4);
  float*  x1    = (float*) alloc(BN * 64 * 4);
  __bf16* xb1   = (__bf16*)alloc(BN * 64 * 2);
  float*  xx1   = (float*) alloc(BN * 4);
  float*  x2    = (float*) alloc(BN * 128 * 4);
  __bf16* xb2   = (__bf16*)alloc(BN * 128 * 2);
  float*  xx2   = (float*) alloc(BN * 4);
  __bf16* wb1   = (__bf16*)alloc(64 * 64 * 2);
  __bf16* wb2   = (__bf16*)alloc(128 * 128 * 2);
  __bf16* wb3   = (__bf16*)alloc(256 * 256 * 2);
  unsigned* gfeat = (unsigned*)alloc(BATCH * 256 * 4);

  const float invcnt = 1.0f / (float)(BN * KNN);
  const dim3 kgrid(NPTS / 64, BATCH);

  // weight + input prep
  prep0_kernel<<<(BN + 255) / 256, 256, 0, stream>>>(x, xb0, xx0);
  wprep_kernel<64, 3, 64><<<(64 * 64 + 255) / 256, 256, 0, stream>>>(w1, wb1);
  wprep_kernel<128, 64, 128><<<(128 * 128 + 255) / 256, 256, 0, stream>>>(w2, wb2);
  wprep_kernel<256, 128, 256><<<(256 * 256 + 255) / 256, 256, 0, stream>>>(w3, wb3);

  // ---- layer 1: 3 -> 64 ----
  zero_kernel<<<2, 256, 0, stream>>>((unsigned*)stats, 512);
  knn_kernel<32><<<kgrid, 64, 0, stream>>>(xb0, xx0, idxb);
  mlp_kernel<3, 64, 64><<<(int)BN, 256, 0, stream>>>(x, idxb, wb1, b1, hmax, stats);
  bnfin_kernel<64, false><<<(int)BN, 64, 0, stream>>>(hmax, stats, g1, be1, x1, xb1, xx1, nullptr, invcnt);

  // ---- layer 2: 64 -> 128 ----
  zero_kernel<<<2, 256, 0, stream>>>((unsigned*)stats, 512);
  knn_kernel<64><<<kgrid, 64, 0, stream>>>(xb1, xx1, idxb);
  mlp_kernel<64, 128, 128><<<(int)BN, 256, 0, stream>>>(x1, idxb, wb2, b2, hmax, stats);
  bnfin_kernel<128, false><<<(int)BN, 128, 0, stream>>>(hmax, stats, g2, be2, x2, xb2, xx2, nullptr, invcnt);

  // ---- layer 3: 128 -> 256 + global max over N ----
  zero_kernel<<<2, 256, 0, stream>>>((unsigned*)stats, 512);
  knn_kernel<128><<<kgrid, 64, 0, stream>>>(xb2, xx2, idxb);
  mlp_kernel<128, 256, 256><<<(int)BN, 256, 0, stream>>>(x2, idxb, wb3, b3, hmax, stats);
  zero_kernel<<<(BATCH * 256 + 255) / 256, 256, 0, stream>>>(gfeat, BATCH * 256);
  bnfin_kernel<256, true><<<(int)BN, 256, 0, stream>>>(hmax, stats, g3, be3, nullptr, nullptr, nullptr, gfeat, invcnt);

  // ---- head ----
  head_kernel<<<(BATCH * 256 + 255) / 256, 256, 0, stream>>>(gfeat, wo, bo, out);
}
